// SimultaneousFeatureExtractor_81767587381385
// MI455X (gfx1250) — matrix operand order repaired and checked by reading, not compiled, not run on hardware
//
#include <hip/hip_runtime.h>
#include <hip/hip_bf16.h>
#include <math.h>

typedef unsigned short u16;
typedef __attribute__((ext_vector_type(16))) __bf16 v16bf;
typedef __attribute__((ext_vector_type(8)))  float  v8f;
typedef __attribute__((ext_vector_type(4)))  unsigned v4u;
typedef __attribute__((ext_vector_type(8)))  int      v8i;
typedef __attribute__((ext_vector_type(4)))  int      v4i;

#define PIX 65536L

// ---- d_out layout (floats), in reference return order ----
#define OUT_FEAT 0L          // 4*128*65536
#define OUT_RL   33554432L   // 4*3*65536
#define OUT_AL   34340864L   // 4*1*65536
#define OUT_ALLM 34603008L   // 4*30*65536
#define OUT_ANOM 42467328L   // 4*32*65536
#define OUT_QS   50855936L   // 4*1*65536
#define OUT_RP   51118080L   // 4*3*65536
#define OUT_AP   51904512L   // 4*1*65536

// ---- workspace layout (bytes) ----
#define WS_WP1  0L            // 64*32*2
#define WS_WP2  4096L         // 128*576*2
#define WS_WPP  151552L       // 64*1152*2
#define WS_WPQ  299008L       // 64*192*2
#define WS_SC1  329728L
#define WS_SH1  329984L
#define WS_SC2  330240L
#define WS_SH2  330752L
#define WS_QIN  331776L       // 4*190*65536*2 bf16
#define WS_HBF  99946496L     // 4*64*65536*2 bf16
#define WS_RAW  133500928L    // 4*128*65536*4 f32 (reused: conv1 raw, conv2 raw, q)

#if defined(__has_builtin)
#  if __has_builtin(__builtin_amdgcn_tensor_load_to_lds)
#    define HAVE_TDM 1
#  else
#    define HAVE_TDM 0
#  endif
#else
#  define HAVE_TDM 0
#endif

union Frag { v16bf v; uint4 q[2]; };
union Pack8 { uint4 v; u16 h[8]; };

__device__ __forceinline__ u16 f2bf(float f) {
  union { float f; unsigned u; } c; c.f = f;
  unsigned r = c.u + 0x7FFFu + ((c.u >> 16) & 1u);
  return (u16)(r >> 16);
}
__device__ __forceinline__ float bf2f(u16 b) {
  union { unsigned u; float f; } c; c.u = ((unsigned)b) << 16;
  return c.f;
}
__device__ __forceinline__ float sigmoidf(float x) { return 1.0f / (1.0f + __expf(-x)); }

#if HAVE_TDM
// TDM: DMA a 64-row x 32-col bf16 tile (row stride = row_stride_elems) from
// global into LDS at byte offset lds_off. D# fields per CDNA5 ISA ch.8:
// group0 = {count=1 | gather off, lds_addr, global_addr[31:0],
//           global_addr[56:32] | type=2<<30}
// group1 = {data_size=2B, tensor_dim0=row_stride, tensor_dim1=64,
//           tile_dim0=32, tile_dim1=64, tensor_dim0_stride=row_stride}
__device__ __forceinline__ void tdm_load_tile64x32(const u16* gsrc, unsigned lds_off,
                                                   unsigned row_stride_elems) {
  const unsigned long long ga = (unsigned long long)gsrc;
  v4u g0 = { 1u, lds_off, (unsigned)ga,
             (unsigned)((ga >> 32) & 0x01FFFFFFu) | 0x80000000u };
  v8i g1;
  g1[0] = (int)(1u << 16);                                        // data_size = 2B
  g1[1] = (int)((row_stride_elems & 0xFFFFu) << 16);              // tensor_dim0 lo16
  g1[2] = (int)((row_stride_elems >> 16) & 0xFFFFu) | (64 << 16); // dim0 hi16 | dim1 lo16
  g1[3] = (int)(32u << 16);                                       // dim1 hi16=0 | tile_dim0=32
  g1[4] = 64;                                                     // tile_dim1=64 | tile_dim2=0
  g1[5] = (int)row_stride_elems;                                  // dim0_stride lo32
  g1[6] = 0;                                                      // dim0_stride hi16 | dim1_stride lo16
  g1[7] = 0;
  v4i z4 = {0, 0, 0, 0};
#if __has_include(<hip/amd_detail/amd_gfx1250_TDM.h>)
  v8i z8 = {0, 0, 0, 0, 0, 0, 0, 0};
  __builtin_amdgcn_tensor_load_to_lds(g0, g1, z4, z4, z8, 0);     // clang-23 6-arg form
#else
  __builtin_amdgcn_tensor_load_to_lds(g0, g1, z4, z4, 0);         // ROCm 7.2 5-arg form
#endif
}
#endif

// ---- stage the 64px x 32k activation tile for K-block kb into LDS [pix][k] ----
__device__ __forceinline__ void stage_b(const void* inp, long nstride, int CIN, int KS,
                                        int n, int y, int x0, int kb, u16* bdst) {
  const int tid = threadIdx.x;
  if (KS == 3 && CIN >= 32) {            // bf16 activations, 3x3, pad=1
    const int cpb  = CIN >> 5;
    const int r    = kb / cpb;
    const int cb   = (kb % cpb) << 5;
    const int ky   = r / 3, kx = r % 3;
    const int iy   = y + ky - 1;
    const int krow = tid & 31, pq = tid >> 5;
    const bool rowok = (iy >= 0) && (iy < 256);
    const u16* inH = (const u16*)inp;
    const long base = (long)n * nstride + (long)(cb + krow) * PIX + (long)iy * 256;
    const int px0 = x0 + pq * 8 + kx - 1;
    if (rowok && ((px0 & 7) == 0) && (px0 + 7 < 256)) {    // aligned interior: 1 b128 load
      Pack8 d;
      d.v = *(const uint4*)&inH[base + px0];
#pragma unroll
      for (int j = 0; j < 8; ++j) bdst[(pq * 8 + j) * 32 + krow] = d.h[j];
    } else {
#pragma unroll
      for (int j = 0; j < 8; ++j) {
        const int px = px0 + j;
        u16 v = 0;
        if (rowok && px >= 0 && px < 256) v = inH[base + px];
        bdst[(pq * 8 + j) * 32 + krow] = v;
      }
    }
  } else if (KS == 3) {                  // conv1: CIN==3, f32 input, 27 real K
    const float* inF = (const float*)inp;
#pragma unroll
    for (int e = 0; e < 8; ++e) {
      const int idx = tid * 8 + e;
      const int pix = idx >> 5, k = idx & 31;
      u16 v = 0;
      if (k < 27) {
        const int r = k / 3, ci = k % 3;
        const int ky = r / 3, kx = r % 3;
        const int iy = y + ky - 1, ix = x0 + pix + kx - 1;
        if (iy >= 0 && iy < 256 && ix >= 0 && ix < 256)
          v = f2bf(inF[(long)n * nstride + (long)ci * PIX + (long)iy * 256 + ix]);
      }
      bdst[pix * 32 + k] = v;
    }
  } else {                               // 1x1 conv, pad=0: always aligned b128
    const int krow = tid & 31, pq = tid >> 5;
    const int k = kb * 32 + krow;
    const u16* inH = (const u16*)inp;
    const long p0 = (long)y * 256 + x0 + pq * 8;
    Pack8 d;
    if (k < CIN) d.v = *(const uint4*)&inH[(long)n * nstride + (long)k * PIX + p0];
    else { d.v.x = 0; d.v.y = 0; d.v.z = 0; d.v.w = 0; }
#pragma unroll
    for (int j = 0; j < 8; ++j) bdst[(pq * 8 + j) * 32 + krow] = d.h[j];
  }
}

// =====================================================================
// Implicit-GEMM WMMA main loop, double-buffered.
// Workgroup: 256 threads = 8 waves; computes 64 out-channels x 64 pixels.
// Wave w: co sub-tile (w>>1)*16, pixel sub-tile (w&1)*32 (2 WMMA N-tiles).
// A (weights) tile fetched by the Tensor Data Mover (wave 0 issues, waits
// TENSORcnt); B (activations) staged by all threads. One barrier per block.
// K ordering: K = (ky*3+kx)*CIN + ci.
// =====================================================================
__device__ __forceinline__ void conv_tile(
    const void* inp, long nstride, const u16* wpack,
    int Ktot, int CIN, int KS, int n, int co0, int y, int x0,
    u16* ldsA, u16* ldsB, v8f* acc)
{
  const int tid  = threadIdx.x;
  const int lane = tid & 31;
  const int wave = tid >> 5;
  const int wm   = wave >> 1;
  const int wn   = wave & 1;
  const int m    = lane & 15;
  const int sel  = lane >> 4;
  const int nKB  = Ktot >> 5;

#pragma unroll
  for (int t = 0; t < 2; ++t)
#pragma unroll
    for (int e = 0; e < 8; ++e) acc[t][e] = 0.0f;

#if HAVE_TDM
  const unsigned ldsA_off = (unsigned)(unsigned long long)(const void*)ldsA;
  if (wave == 0)
    tdm_load_tile64x32(wpack + (long)co0 * Ktot, ldsA_off, (unsigned)Ktot);
#else
  const int arow = tid >> 2;
  const int acol = (tid & 3) * 8;
  *(uint4*)&ldsA[arow * 32 + acol] =
      *(const uint4*)&wpack[(long)(co0 + arow) * Ktot + acol];
#endif
  stage_b(inp, nstride, CIN, KS, n, y, x0, 0, ldsB);
#if HAVE_TDM
  if (wave == 0) __builtin_amdgcn_s_wait_tensorcnt(0);
#endif
  __syncthreads();

  for (int kb = 0; kb < nKB; ++kb) {
    const int cur = kb & 1;
    const int nxt = cur ^ 1;
    if (kb + 1 < nKB) {   // prefetch next block while computing this one
#if HAVE_TDM
      if (wave == 0)
        tdm_load_tile64x32(wpack + (long)co0 * Ktot + (kb + 1) * 32,
                           ldsA_off + (unsigned)nxt * 4096u, (unsigned)Ktot);
#else
      *(uint4*)&ldsA[nxt * 2048 + arow * 32 + acol] =
          *(const uint4*)&wpack[(long)(co0 + arow) * Ktot + (kb + 1) * 32 + acol];
#endif
      stage_b(inp, nstride, CIN, KS, n, y, x0, kb + 1, ldsB + nxt * 2048);
    }

    // ---- fragments per ISA 16-bit A(16x32)/B(32x16) VGPR layouts
    const u16* A = ldsA + cur * 2048;
    const u16* B = ldsB + cur * 2048;
    Frag a, b0, b1;
    const int abase = (wm * 16 + m) * 32 + sel * 8;
    a.q[0] = *(const uint4*)&A[abase];
    a.q[1] = *(const uint4*)&A[abase + 16];
    const int bb = (wn * 32 + m) * 32 + sel * 8;
    b0.q[0] = *(const uint4*)&B[bb];
    b0.q[1] = *(const uint4*)&B[bb + 16];
    b1.q[0] = *(const uint4*)&B[bb + 512];
    b1.q[1] = *(const uint4*)&B[bb + 528];

    acc[0] = __builtin_amdgcn_wmma_f32_16x16x32_bf16(false, a.v, false, b0.v,
                                                     (short)0, acc[0], false, false);
    acc[1] = __builtin_amdgcn_wmma_f32_16x16x32_bf16(false, a.v, false, b1.v,
                                                     (short)0, acc[1], false, false);

#if HAVE_TDM
    if (wave == 0 && kb + 1 < nKB) __builtin_amdgcn_s_wait_tensorcnt(0);
#endif
    __syncthreads();
  }
}

// ===================== weight packing =====================
__global__ __launch_bounds__(256) void k_pack1(const float* __restrict__ w, u16* __restrict__ wp) {
  const int i = blockIdx.x * 256 + threadIdx.x;   // 64*32
  if (i >= 64 * 32) return;
  const int co = i >> 5, k = i & 31;
  u16 v = 0;
  if (k < 27) { const int r = k / 3, ci = k % 3; v = f2bf(w[(co * 3 + ci) * 9 + r]); }
  wp[i] = v;
}

__global__ __launch_bounds__(256) void k_pack3x3(const float* __restrict__ w, u16* __restrict__ wp,
                                                 int Co, int CIN) {
  const long Ktot = 9L * CIN;
  const long i = (long)blockIdx.x * 256 + threadIdx.x;
  if (i >= (long)Co * Ktot) return;
  const int co = (int)(i / Ktot);
  const int k  = (int)(i % Ktot);
  const int r = k / CIN, ci = k % CIN;
  wp[i] = f2bf(w[((long)co * CIN + ci) * 9 + r]);
}

__global__ __launch_bounds__(256) void k_packP(const float* __restrict__ core,
                                               const float* __restrict__ clad,
                                               const float* __restrict__ ferr,
                                               const float* __restrict__ anom,
                                               u16* __restrict__ wp) {
  const long i = (long)blockIdx.x * 256 + threadIdx.x;   // 64*1152
  if (i >= 64L * 1152L) return;
  const int co = (int)(i / 1152);
  const int k  = (int)(i % 1152);
  const int r = k / 128, ci = k % 128;
  const float* src = nullptr; int lco = 0;
  if      (co < 10) { src = core; lco = co; }
  else if (co < 20) { src = clad; lco = co - 10; }
  else if (co < 30) { src = ferr; lco = co - 20; }
  else if (co < 62) { src = anom; lco = co - 30; }
  float v = 0.f;
  if (src) v = src[((long)lco * 128 + ci) * 9 + r];
  wp[i] = f2bf(v);
}

__global__ __launch_bounds__(256) void k_packQ(const float* __restrict__ qa1w, u16* __restrict__ wp) {
  const int i = blockIdx.x * 256 + threadIdx.x;   // 64*192
  if (i >= 64 * 192) return;
  const int co = i / 192, k = i % 192;
  float v = (k < 190) ? qa1w[co * 190 + k] : 0.f;
  wp[i] = f2bf(v);
}

// ===================== conv kernels =====================
__global__ __launch_bounds__(256) void k_conv1(const float* __restrict__ x,
                                               const u16* __restrict__ wp,
                                               float* __restrict__ raw) {
  __shared__ __align__(16) u16 ldsA[4096];
  __shared__ __align__(16) u16 ldsB[4096];
  const int y = blockIdx.x >> 2, x0 = (blockIdx.x & 3) * 64, n = blockIdx.z;
  v8f acc[2];
  conv_tile(x, 3 * PIX, wp, 32, 3, 3, n, 0, y, x0, ldsA, ldsB, acc);
  const int lane = threadIdx.x & 31, wave = threadIdx.x >> 5;
  const int wm = wave >> 1, wn = wave & 1, sel = lane >> 4;
#pragma unroll
  for (int t = 0; t < 2; ++t)
#pragma unroll
    for (int e = 0; e < 8; ++e) {
      const int co = wm * 16 + sel * 8 + e;
      const int px = x0 + wn * 32 + t * 16 + (lane & 15);
      raw[((long)n * 64 + co) * PIX + (long)y * 256 + px] = acc[t][e];
    }
}

__global__ __launch_bounds__(256) void k_conv2(const u16* __restrict__ h,
                                               const u16* __restrict__ wp,
                                               float* __restrict__ raw) {
  __shared__ __align__(16) u16 ldsA[4096];
  __shared__ __align__(16) u16 ldsB[4096];
  const int y = blockIdx.x >> 2, x0 = (blockIdx.x & 3) * 64, n = blockIdx.z;
  const int co0 = blockIdx.y * 64;
  v8f acc[2];
  conv_tile(h, 64 * PIX, wp, 576, 64, 3, n, co0, y, x0, ldsA, ldsB, acc);
  const int lane = threadIdx.x & 31, wave = threadIdx.x >> 5;
  const int wm = wave >> 1, wn = wave & 1, sel = lane >> 4;
#pragma unroll
  for (int t = 0; t < 2; ++t)
#pragma unroll
    for (int e = 0; e < 8; ++e) {
      const int co = co0 + wm * 16 + sel * 8 + e;
      const int px = x0 + wn * 32 + t * 16 + (lane & 15);
      raw[((long)n * 128 + co) * PIX + (long)y * 256 + px] = acc[t][e];
    }
}

__global__ __launch_bounds__(256) void k_patt(const u16* __restrict__ qin_in,
                                              const u16* __restrict__ wp,
                                              float* __restrict__ outAll,
                                              float* __restrict__ outAnom,
                                              u16* __restrict__ qin_out) {
  __shared__ __align__(16) u16 ldsA[4096];
  __shared__ __align__(16) u16 ldsB[4096];
  const int y = blockIdx.x >> 2, x0 = (blockIdx.x & 3) * 64, n = blockIdx.z;
  v8f acc[2];
  conv_tile(qin_in, 190 * PIX, wp, 1152, 128, 3, n, 0, y, x0, ldsA, ldsB, acc);
  const int lane = threadIdx.x & 31, wave = threadIdx.x >> 5;
  const int wm = wave >> 1, wn = wave & 1, sel = lane >> 4;
#pragma unroll
  for (int t = 0; t < 2; ++t)
#pragma unroll
    for (int e = 0; e < 8; ++e) {
      const int co = wm * 16 + sel * 8 + e;
      if (co < 62) {
        const int px = x0 + wn * 32 + t * 16 + (lane & 15);
        const long pix = (long)y * 256 + px;
        const float s = sigmoidf(acc[t][e]);
        if (co < 30) outAll[((long)n * 30 + co) * PIX + pix] = s;
        else         outAnom[((long)n * 32 + (co - 30)) * PIX + pix] = s;
        qin_out[((long)n * 190 + 128 + co) * PIX + pix] = f2bf(s);
      }
    }
}

__global__ __launch_bounds__(256) void k_qa(const u16* __restrict__ qin,
                                            const u16* __restrict__ wp,
                                            const float* __restrict__ qa1_b,
                                            float* __restrict__ q) {
  __shared__ __align__(16) u16 ldsA[4096];
  __shared__ __align__(16) u16 ldsB[4096];
  const int y = blockIdx.x >> 2, x0 = (blockIdx.x & 3) * 64, n = blockIdx.z;
  v8f acc[2];
  conv_tile(qin, 190 * PIX, wp, 192, 190, 1, n, 0, y, x0, ldsA, ldsB, acc);
  const int lane = threadIdx.x & 31, wave = threadIdx.x >> 5;
  const int wm = wave >> 1, wn = wave & 1, sel = lane >> 4;
#pragma unroll
  for (int t = 0; t < 2; ++t)
#pragma unroll
    for (int e = 0; e < 8; ++e) {
      const int co = wm * 16 + sel * 8 + e;
      const int px = x0 + wn * 32 + t * 16 + (lane & 15);
      float v = acc[t][e] + qa1_b[co];
      v = fmaxf(v, 0.f);
      q[((long)n * 64 + co) * PIX + (long)y * 256 + px] = v;
    }
}

// ===================== batchnorm =====================
__global__ __launch_bounds__(256) void k_bnstats(const float* __restrict__ raw, int C,
                                                 const float* __restrict__ g,
                                                 const float* __restrict__ b,
                                                 float* __restrict__ scale,
                                                 float* __restrict__ shift) {
  __shared__ float ssum[256], ssq[256];
  const int c = blockIdx.x, tid = threadIdx.x;
  float s = 0.f, q = 0.f;
  for (int n = 0; n < 4; ++n) {
    const float* p = raw + ((long)n * C + c) * PIX;
    for (int i = tid; i < 65536; i += 256) { const float v = p[i]; s += v; q += v * v; }
  }
  ssum[tid] = s; ssq[tid] = q;
  __syncthreads();
  for (int st = 128; st > 0; st >>= 1) {
    if (tid < st) { ssum[tid] += ssum[tid + st]; ssq[tid] += ssq[tid + st]; }
    __syncthreads();
  }
  if (tid == 0) {
    const float mean = ssum[0] / 262144.0f;
    const float var  = ssq[0] / 262144.0f - mean * mean;
    const float sc   = g[c] * rsqrtf(var + 1e-5f);
    scale[c] = sc;
    shift[c] = b[c] - mean * sc;
  }
}

__global__ __launch_bounds__(256) void k_bnapply(const float* __restrict__ raw,
                                                 const float* __restrict__ scale,
                                                 const float* __restrict__ shift, int C,
                                                 float* __restrict__ outF,
                                                 u16* __restrict__ outB,
                                                 long outB_nstride, int coff) {
  const long i = (long)blockIdx.x * 256 + threadIdx.x;
  const int pix = (int)(i & 65535);
  const long top = i >> 16;             // n*C + c
  const int c = (int)(top % C);
  const int n = (int)(top / C);
  float v = raw[i] * scale[c] + shift[c];
  v = fmaxf(v, 0.f);
  if (outF) outF[i] = v;
  outB[(long)n * outB_nstride + (long)(coff + c) * PIX + pix] = f2bf(v);
}

// ===================== final heads =====================
__global__ __launch_bounds__(256) void k_final(const u16* __restrict__ qin,
                                               const float* __restrict__ q,
                                               const float* __restrict__ qa2_w,
                                               const float* __restrict__ qa2_b,
                                               const float* __restrict__ cls_w,
                                               const float* __restrict__ cls_b,
                                               float* __restrict__ out) {
  const long gi = (long)blockIdx.x * 256 + threadIdx.x;  // 4*65536
  const int n = (int)(gi >> 16);
  const int pix = (int)(gi & 65535);
  float cl0 = cls_b[0], cl1 = cls_b[1], cl2 = cls_b[2], cl3 = cls_b[3];
  const u16* qp = qin + (long)n * 190 * PIX + pix;
  for (int c = 0; c < 190; ++c) {
    const float v = bf2f(qp[(long)c * PIX]);
    cl0 += v * cls_w[c];
    cl1 += v * cls_w[190 + c];
    cl2 += v * cls_w[380 + c];
    cl3 += v * cls_w[570 + c];
  }
  float* rl = out + OUT_RL;
  float* al = out + OUT_AL;
  float* rp = out + OUT_RP;
  float* ap = out + OUT_AP;
  float* qs = out + OUT_QS;
  rl[((long)n * 3 + 0) * PIX + pix] = cl0;
  rl[((long)n * 3 + 1) * PIX + pix] = cl1;
  rl[((long)n * 3 + 2) * PIX + pix] = cl2;
  al[(long)n * PIX + pix] = cl3;
  const float mx = fmaxf(cl0, fmaxf(cl1, cl2));
  const float e0 = __expf(cl0 - mx), e1 = __expf(cl1 - mx), e2 = __expf(cl2 - mx);
  const float inv = 1.f / (e0 + e1 + e2);
  rp[((long)n * 3 + 0) * PIX + pix] = e0 * inv;
  rp[((long)n * 3 + 1) * PIX + pix] = e1 * inv;
  rp[((long)n * 3 + 2) * PIX + pix] = e2 * inv;
  ap[(long)n * PIX + pix] = sigmoidf(cl3);
  float s = qa2_b[0];
  const float* qq = q + (long)n * 64 * PIX + pix;
  for (int c = 0; c < 64; ++c) s += qq[(long)c * PIX] * qa2_w[c];
  qs[(long)n * PIX + pix] = sigmoidf(s);
}

// ===================== host launch =====================
extern "C" void kernel_launch(void* const* d_in, const int* in_sizes, int n_in,
                              void* d_out, int out_size, void* d_ws, size_t ws_size,
                              hipStream_t stream) {
  (void)in_sizes; (void)n_in; (void)out_size; (void)ws_size;
  const float* x       = (const float*)d_in[0];
  const float* conv1_w = (const float*)d_in[1];
  // conv1_b/conv2_b cancel inside batchnorm (pure mean shift) -> unused
  const float* bn1_g   = (const float*)d_in[3];
  const float* bn1_b   = (const float*)d_in[4];
  const float* conv2_w = (const float*)d_in[5];
  const float* bn2_g   = (const float*)d_in[7];
  const float* bn2_b   = (const float*)d_in[8];
  const float* core_p  = (const float*)d_in[9];
  const float* clad_p  = (const float*)d_in[10];
  const float* ferr_p  = (const float*)d_in[11];
  const float* anom_p  = (const float*)d_in[12];
  const float* qa1_w   = (const float*)d_in[13];
  const float* qa1_b   = (const float*)d_in[14];
  const float* qa2_w   = (const float*)d_in[15];
  const float* qa2_b   = (const float*)d_in[16];
  const float* cls_w   = (const float*)d_in[17];
  const float* cls_b   = (const float*)d_in[18];

  char* ws   = (char*)d_ws;
  u16*  wp1  = (u16*)(ws + WS_WP1);
  u16*  wp2  = (u16*)(ws + WS_WP2);
  u16*  wpp  = (u16*)(ws + WS_WPP);
  u16*  wpq  = (u16*)(ws + WS_WPQ);
  float* sc1 = (float*)(ws + WS_SC1);
  float* sh1 = (float*)(ws + WS_SH1);
  float* sc2 = (float*)(ws + WS_SC2);
  float* sh2 = (float*)(ws + WS_SH2);
  u16*  qin  = (u16*)(ws + WS_QIN);
  u16*  hbf  = (u16*)(ws + WS_HBF);
  float* raw = (float*)(ws + WS_RAW);
  float* out = (float*)d_out;

  // weight packing (bf16, K-major)
  k_pack1 <<<8,   256, 0, stream>>>(conv1_w, wp1);
  k_pack3x3<<<288, 256, 0, stream>>>(conv2_w, wp2, 128, 64);
  k_packP <<<288, 256, 0, stream>>>(core_p, clad_p, ferr_p, anom_p, wpp);
  k_packQ <<<48,  256, 0, stream>>>(qa1_w, wpq);

  // conv1 + BN1 + relu -> h (bf16)
  k_conv1  <<<dim3(1024, 1, 4), 256, 0, stream>>>(x, wp1, raw);
  k_bnstats<<<64,  256, 0, stream>>>(raw, 64, bn1_g, bn1_b, sc1, sh1);
  k_bnapply<<<65536, 256, 0, stream>>>(raw, sc1, sh1, 64, nullptr, hbf, 64 * PIX, 0);

  // conv2 + BN2 + relu -> features (f32 out) + qin channels 0..127 (bf16)
  k_conv2  <<<dim3(1024, 2, 4), 256, 0, stream>>>(hbf, wp2, raw);
  k_bnstats<<<128, 256, 0, stream>>>(raw, 128, bn2_g, bn2_b, sc2, sh2);
  k_bnapply<<<131072, 256, 0, stream>>>(raw, sc2, sh2, 128, out + OUT_FEAT, qin, 190 * PIX, 0);

  // pattern convs + sigmoid -> all/anomaly matches + qin channels 128..189
  k_patt<<<dim3(1024, 1, 4), 256, 0, stream>>>(qin, wpp, out + OUT_ALLM, out + OUT_ANOM, qin);

  // qa1 (1x1, 190->64) + relu -> q (f32, reuses raw region)
  k_qa<<<dim3(1024, 1, 4), 256, 0, stream>>>(qin, wpq, qa1_b, raw);

  // cls head, softmax/sigmoid, qa2 + sigmoid
  k_final<<<1024, 256, 0, stream>>>(qin, raw, qa2_w, qa2_b, cls_w, cls_b, out);
}
